// Network_50087908606125
// MI455X (gfx1250) — matrix operand order repaired and checked by reading, not compiled, not run on hardware
//
#include <hip/hip_runtime.h>
#include <hip/hip_bf16.h>
#include <math.h>

// ---------------------------------------------------------------------------
// e3nn-style equivariant GNN forward for MI455X (gfx1250, wave32, WMMA).
//
// The per-edge radial MLPs dominate FLOPs ([E,256]x[256,272] and
// [E,256]x[256,304] ~ 177 GFLOP) and run on v_wmma_f32_16x16x32_f16, fused
// with the Clebsch-Gordan tensor products and the atomic segment-sum scatter
// so the 272/304 per-edge weights never touch HBM. 32 edges (2 M-tiles) per
// 256-thread block so each B fragment load feeds 2 WMMAs (halves L2 weight
// traffic vs 16-edge tiles); next n-tile's B fragments are prefetched.
// ---------------------------------------------------------------------------

typedef __attribute__((ext_vector_type(16))) _Float16 v16h;
typedef __attribute__((ext_vector_type(8)))  _Float16 v8h;
typedef __attribute__((ext_vector_type(8)))  float    v8f;

#define RELU_NORM       1.4142135623730951f
#define TANH_NORM       1.5927315f
#define INV_SQRT_NEIGH  0.5129891760425771f   // 1/sqrt(3.8)
#define SQRT_HALF       0.7071067811865476f
#define FINAL_B         0.14433756729740643f  // sqrt(1/48)
#define INV_SQRT3       0.5773502691896258f
#define INV16           0.0625f               // 1/sqrt(256)

// CG (real Wigner-3j) table float offsets, in order:
// (000)(110)(220)(330)(011)(101)(121)(211)(231)(321)(111)(221)(331)
#define OFF000 0
#define OFF110 1
#define OFF220 10
#define OFF330 35
#define OFF011 84
#define OFF101 93
#define OFF121 102
#define OFF211 147
#define OFF231 192
#define OFF321 297
#define OFF111 402
#define OFF221 429
#define OFF331 504
#define CG_FLOATS 651

// ---------------------------------------------------------------------------
// Exact real Wigner-3j computed on device (1 thread per table, runs once).
// ---------------------------------------------------------------------------
__device__ double dfact(int n) { double r = 1.0; for (int i = 2; i <= n; ++i) r *= (double)i; return r; }

__device__ void real_change(int l, double Ar[7][7], double Ai[7][7]) {
  for (int i = 0; i < 7; ++i) for (int j = 0; j < 7; ++j) { Ar[i][j] = 0.0; Ai[i][j] = 0.0; }
  Ar[l][l] = 1.0;
  const double s = 0.7071067811865476;
  for (int m = 1; m <= l; ++m) {
    double pm = (m & 1) ? -1.0 : 1.0;
    Ar[l + m][l - m] = s;  Ar[l + m][l + m] = pm * s;
    Ai[l - m][l - m] = s;  Ai[l - m][l + m] = -pm * s;
  }
}

__device__ void compute_real_w3j(int l1, int l2, int l3, float* out) {
  const int d1 = 2 * l1 + 1, d2 = 2 * l2 + 1, d3 = 2 * l3 + 1;
  double C[7][7][3];
  for (int a = 0; a < 7; ++a) for (int b = 0; b < 7; ++b) for (int c = 0; c < 3; ++c) C[a][b][c] = 0.0;
  for (int m1 = -l1; m1 <= l1; ++m1)
    for (int m2 = -l2; m2 <= l2; ++m2) {
      int m3 = m1 + m2;
      if (m3 < -l3 || m3 > l3) continue;
      double pref = sqrt((double)(2 * l3 + 1) * dfact(l3 + l1 - l2) * dfact(l3 - l1 + l2) *
                         dfact(l1 + l2 - l3) / dfact(l1 + l2 + l3 + 1));
      pref *= sqrt(dfact(l3 + m3) * dfact(l3 - m3) * dfact(l1 - m1) * dfact(l1 + m1) *
                   dfact(l2 - m2) * dfact(l2 + m2));
      double s = 0.0;
      for (int k = 0; k <= l1 + l2 - l3; ++k) {
        int d[6] = {k, l1 + l2 - l3 - k, l1 - m1 - k, l2 + m2 - k, l3 - l2 + m1 + k, l3 - l1 - m2 + k};
        bool ok = true;
        for (int q = 0; q < 6; ++q) if (d[q] < 0) ok = false;
        if (!ok) continue;
        double den = 1.0;
        for (int q = 0; q < 6; ++q) den *= dfact(d[q]);
        s += ((k & 1) ? -1.0 : 1.0) / den;
      }
      C[m1 + l1][m2 + l2][m3 + l3] = pref * s;
    }
  // w3j: w[a][b][c] = (-1)^(l1-l2-m3)/sqrt(2l3+1) * C[a][b][l3-m3], m3 = c-l3
  double w[7][7][3];
  for (int a = 0; a < d1; ++a)
    for (int b = 0; b < d2; ++b)
      for (int c = 0; c < d3; ++c) {
        int m3 = c - l3;
        double sgn = ((l1 - l2 - m3) & 1) ? -1.0 : 1.0;
        w[a][b][c] = sgn / sqrt((double)(2 * l3 + 1)) * C[a][b][l3 - m3];
      }
  double A1r[7][7], A1i[7][7], A2r[7][7], A2i[7][7], A3r[7][7], A3i[7][7];
  real_change(l1, A1r, A1i); real_change(l2, A2r, A2i); real_change(l3, A3r, A3i);
  double Tre[7][7][3], Tim[7][7][3];
  double nr = 0.0, ni = 0.0;
  for (int i = 0; i < d1; ++i)
    for (int j = 0; j < d2; ++j)
      for (int k = 0; k < d3; ++k) {
        double tr = 0.0, ti = 0.0;
        for (int a = 0; a < d1; ++a) {
          double a1r = A1r[i][a], a1i = A1i[i][a];
          if (a1r == 0.0 && a1i == 0.0) continue;
          for (int b = 0; b < d2; ++b) {
            double a2r = A2r[j][b], a2i = A2i[j][b];
            if (a2r == 0.0 && a2i == 0.0) continue;
            double pr = a1r * a2r - a1i * a2i;
            double pi = a1r * a2i + a1i * a2r;
            for (int c = 0; c < d3; ++c) {
              double wv = w[a][b][c];
              if (wv == 0.0) continue;
              double a3r = A3r[k][c], a3i = A3i[k][c];
              tr += (pr * a3r - pi * a3i) * wv;
              ti += (pr * a3i + pi * a3r) * wv;
            }
          }
        }
        Tre[i][j][k] = tr; Tim[i][j][k] = ti;
        nr += tr * tr; ni += ti * ti;
      }
  for (int i = 0; i < d1; ++i)
    for (int j = 0; j < d2; ++j)
      for (int k = 0; k < d3; ++k)
        out[(i * d2 + j) * d3 + k] = (float)((nr >= ni) ? Tre[i][j][k] : Tim[i][j][k]);
}

__global__ void cg_prep_kernel(float* __restrict__ cg) {
  const int L1[13] = {0, 1, 2, 3, 0, 1, 1, 2, 2, 3, 1, 2, 3};
  const int L2[13] = {0, 1, 2, 3, 1, 0, 2, 1, 3, 2, 1, 2, 3};
  const int L3[13] = {0, 0, 0, 0, 1, 1, 1, 1, 1, 1, 1, 1, 1};
  const int OFF[13] = {OFF000, OFF110, OFF220, OFF330, OFF011, OFF101, OFF121,
                       OFF211, OFF231, OFF321, OFF111, OFF221, OFF331};
  int t = threadIdx.x;
  if (t < 13) compute_real_w3j(L1[t], L2[t], L3[t], cg + OFF[t]);
}

// ---------------------------------------------------------------------------
// Repack w2c/w2f (f32 row-major [256][cols]) into WMMA B-fragment order,
// fold 1/sqrt(256). B k order used here (contiguous, matches A): lanes 0-15 hold K=k0..k0+15,
// lanes 16-31 hold K=k0+16..k0+31; col = lane&15; 16 f16 contiguous per lane.
// ---------------------------------------------------------------------------
__global__ void wprep_kernel(const float* __restrict__ w2c, const float* __restrict__ w2f,
                             _Float16* __restrict__ Bc, _Float16* __restrict__ Bf) {
  int i = blockIdx.x * blockDim.x + threadIdx.x;
  const int nc = 17 * 8 * 32 * 16;  // 69632
  const int nf = 19 * 8 * 32 * 16;  // 77824
  if (i < nc) {
    int e = i & 15, lane = (i >> 4) & 31, k = (i >> 9) & 7, nt = i >> 12;
    int K = k * 32 + (lane >> 4) * 16 + e;
    int col = nt * 16 + (lane & 15);
    Bc[i] = (_Float16)(w2c[K * 272 + col] * INV16);
  }
  if (i < nf) {
    int e = i & 15, lane = (i >> 4) & 31, k = (i >> 9) & 7, nt = i >> 12;
    int K = k * 32 + (lane >> 4) * 16 + e;
    int col = nt * 16 + (lane & 15);
    Bf[i] = (_Float16)(w2f[K * 304 + col] * INV16);
  }
}

// ---------------------------------------------------------------------------
// Per-edge geometry: vec, length, real SH l=0..3, radial gaussians; scatter
// SH into node features x0 (raw sum; 1/sqrt(3.8) folded at gather).
// ---------------------------------------------------------------------------
__global__ __launch_bounds__(256) void edge_geom_kernel(
    const float* __restrict__ pos, const int* __restrict__ esrc, const int* __restrict__ edst,
    float* __restrict__ sh, float* __restrict__ emb, float* __restrict__ x0, int E) {
  int e = blockIdx.x * blockDim.x + threadIdx.x;
  if (e >= E) return;
  int s = esrc[e], d = edst[e];
  float vx = pos[s * 3 + 0] - pos[d * 3 + 0];
  float vy = pos[s * 3 + 1] - pos[d * 3 + 1];
  float vz = pos[s * 3 + 2] - pos[d * 3 + 2];
  float len = sqrtf(vx * vx + vy * vy + vz * vz);
  float inv = 1.0f / fmaxf(len, 1e-9f);
  float x = vx * inv, y = vy * inv, z = vz * inv;
  float x2 = x * x, y2 = y * y, z2 = z * z;
  float S[16];
  S[0] = 1.0f;
  S[1] = 1.7320508f * y;  S[2] = 1.7320508f * z;  S[3] = 1.7320508f * x;
  S[4] = 3.8729833f * x * y; S[5] = 3.8729833f * y * z;
  S[6] = 1.1180340f * (3.0f * z2 - 1.0f);
  S[7] = 3.8729833f * x * z; S[8] = 1.9364917f * (x2 - y2);
  S[9]  = 2.0916501f * y * (3.0f * x2 - y2);
  S[10] = 10.2469508f * x * y * z;
  S[11] = 1.6201852f * y * (5.0f * z2 - 1.0f);
  S[12] = 1.3228757f * z * (5.0f * z2 - 3.0f);
  S[13] = 1.6201852f * x * (5.0f * z2 - 1.0f);
  S[14] = 5.1234754f * z * (x2 - y2);
  S[15] = 2.0916501f * x * (x2 - 3.0f * y2);
#pragma unroll
  for (int c = 0; c < 16; ++c) {
    sh[e * 16 + c] = S[c];
    atomicAdd(&x0[d * 16 + c], S[c]);
  }
  const float ctr[3] = {1.0f, 1.5f, 2.0f};
#pragma unroll
  for (int j = 0; j < 3; ++j) {
    float t = (len - ctr[j]) * 2.0f;  // /0.5
    emb[e * 3 + j] = __expf(-t * t) * 1.7320508f;
  }
}

// ---------------------------------------------------------------------------
// WMMA GEMM for a 32-edge tile (2 M-tiles): D[32][NTILES*16] = hA[32][256] x B.
// A k order used here (contiguous, matches B packing): lane 0-15 -> row=lane, K in {k0..k0+15};
// lanes 16-31 -> row=lane-16, K in {k0+16..k0+31}.
// Each B fragment (global, fragment-packed) feeds both M-tiles' WMMAs.
// ---------------------------------------------------------------------------
template <int NTILES, int STRIDE>
__device__ __forceinline__ void fc_gemm(const _Float16* __restrict__ Bfrag,
                                        const _Float16* __restrict__ hA,  // LDS 32x256
                                        float* __restrict__ sW,           // LDS 32xSTRIDE
                                        int tid) {
  int wv = tid >> 5, lane = tid & 31;
  int row = lane & 15, kh = lane >> 4;
  for (int nt = wv; nt < NTILES; nt += 8) {
    // keep the weight stream ahead of the matrix pipe (global_prefetch_b8)
    if (nt + 8 < NTILES)
      __builtin_prefetch(Bfrag + (size_t)((nt + 8) * 8 * 32) * 16, 0, 1);
    v8f acc0 = {};
    v8f acc1 = {};
#pragma unroll
    for (int k = 0; k < 8; ++k) {
      v16h b = *(const v16h*)(Bfrag + (size_t)(((nt * 8 + k) * 32) + lane) * 16);
      const v8h a00 = *(const v8h*)(hA + row * 256 + k * 32 + kh * 16);
      const v8h a01 = *(const v8h*)(hA + row * 256 + k * 32 + 8 + kh * 16);
      v16h a0 = __builtin_shufflevector(a00, a01, 0, 1, 2, 3, 4, 5, 6, 7,
                                        8, 9, 10, 11, 12, 13, 14, 15);
      const v8h a10 = *(const v8h*)(hA + (16 + row) * 256 + k * 32 + kh * 16);
      const v8h a11 = *(const v8h*)(hA + (16 + row) * 256 + k * 32 + 8 + kh * 16);
      v16h a1 = __builtin_shufflevector(a10, a11, 0, 1, 2, 3, 4, 5, 6, 7,
                                        8, 9, 10, 11, 12, 13, 14, 15);
      acc0 = __builtin_amdgcn_wmma_f32_16x16x32_f16(false, a0, false, b, (short)0, acc0,
                                                    false, false);
      acc1 = __builtin_amdgcn_wmma_f32_16x16x32_f16(false, a1, false, b, (short)0, acc1,
                                                    false, false);
    }
#pragma unroll
    for (int r = 0; r < 8; ++r) {
      sW[(r + kh * 8) * STRIDE + nt * 16 + row] = acc0[r];        // M = r + kh*8
      sW[(16 + r + kh * 8) * STRIDE + nt * 16 + row] = acc1[r];   // M-tile 1
    }
  }
}

// ---------------------------------------------------------------------------
// Fused conv layer: radial MLP (WMMA) -> FullyConnectedTensorProduct
// (x[src] (x) sh with per-edge weights) -> atomic scatter into x1[dst][160].
// Block = 256 threads (8 waves), one 32-edge tile (2 M-tiles) per block.
// ---------------------------------------------------------------------------
__global__ __launch_bounds__(256) void conv_tp_kernel(
    const float* __restrict__ sh, const float* __restrict__ emb, const float* __restrict__ x0,
    const int* __restrict__ esrc, const int* __restrict__ edst, const float* __restrict__ w1c,
    const _Float16* __restrict__ Bc, const float* __restrict__ cg,
    float* __restrict__ x1, int E) {
  __shared__ float sW1[768];
  __shared__ float sEMB[32][4];
  __shared__ __align__(32) _Float16 hA[32 * 256];
  __shared__ float sSH[32][16];
  __shared__ float sX[32][16];
  __shared__ float sWc[32][272];
  __shared__ float sT[32][32];

  int tid = threadIdx.x;
  int e0 = blockIdx.x * 32;

  for (int i = tid; i < 768; i += 256) sW1[i] = w1c[i];
  if (tid < 96) {
    int r = tid / 3, c = tid % 3;
    int ee = e0 + r; if (ee >= E) ee = E - 1;
    sEMB[r][c] = emb[ee * 3 + c];
  }
  for (int i = tid; i < 512; i += 256) {
    int r = i >> 4, c = i & 15;
    int ee = e0 + r; if (ee >= E) ee = E - 1;
    sSH[r][c] = sh[ee * 16 + c];
    sX[r][c] = x0[esrc[ee] * 16 + c] * INV_SQRT_NEIGH;
  }
  __syncthreads();

  // hidden layer: relu(emb @ w1/sqrt(3)) * RELU_NORM, in f16 for WMMA A
  for (int i = tid; i < 8192; i += 256) {
    int r = i >> 8, c = i & 255;
    float v = (sEMB[r][0] * sW1[c] + sEMB[r][1] * sW1[256 + c] + sEMB[r][2] * sW1[512 + c]) * INV_SQRT3;
    hA[r * 256 + c] = (_Float16)(fmaxf(v, 0.0f) * RELU_NORM);
  }
  __syncthreads();

  fc_gemm<17, 272>(Bc, hA, &sWc[0][0], tid);
  __syncthreads();

  // per-edge CG contractions (t-vectors), one thread per edge
  if (tid < 32) {
    const float* xv = &sX[tid][0];
    const float* sv = &sSH[tid][0];
    float* T = &sT[tid][0];
    float t;
    t = xv[0] * sv[0] * cg[OFF000]; T[0] = t;
    t = 0.f; for (int i = 0; i < 3; ++i) for (int j = 0; j < 3; ++j) t += xv[1 + i] * sv[1 + j] * cg[OFF110 + i * 3 + j]; T[1] = t;
    t = 0.f; for (int i = 0; i < 5; ++i) for (int j = 0; j < 5; ++j) t += xv[4 + i] * sv[4 + j] * cg[OFF220 + i * 5 + j]; T[2] = t;
    t = 0.f; for (int i = 0; i < 7; ++i) for (int j = 0; j < 7; ++j) t += xv[9 + i] * sv[9 + j] * cg[OFF330 + i * 7 + j]; T[3] = t;
    for (int k = 0; k < 3; ++k) {
      float v;
      v = 0.f; for (int j = 0; j < 3; ++j) v += xv[0] * sv[1 + j] * cg[OFF011 + j * 3 + k]; T[4 + k] = v;
      v = 0.f; for (int i = 0; i < 3; ++i) v += xv[1 + i] * sv[0] * cg[OFF101 + i * 3 + k]; T[7 + k] = v;
      v = 0.f; for (int i = 0; i < 3; ++i) for (int j = 0; j < 5; ++j) v += xv[1 + i] * sv[4 + j] * cg[OFF121 + (i * 5 + j) * 3 + k]; T[10 + k] = v;
      v = 0.f; for (int i = 0; i < 5; ++i) for (int j = 0; j < 3; ++j) v += xv[4 + i] * sv[1 + j] * cg[OFF211 + (i * 3 + j) * 3 + k]; T[13 + k] = v;
      v = 0.f; for (int i = 0; i < 5; ++i) for (int j = 0; j < 7; ++j) v += xv[4 + i] * sv[9 + j] * cg[OFF231 + (i * 7 + j) * 3 + k]; T[16 + k] = v;
      v = 0.f; for (int i = 0; i < 7; ++i) for (int j = 0; j < 5; ++j) v += xv[9 + i] * sv[4 + j] * cg[OFF321 + (i * 5 + j) * 3 + k]; T[19 + k] = v;
      v = 0.f; for (int i = 0; i < 3; ++i) for (int j = 0; j < 3; ++j) v += xv[1 + i] * sv[1 + j] * cg[OFF111 + (i * 3 + j) * 3 + k]; T[22 + k] = v;
      v = 0.f; for (int i = 0; i < 5; ++i) for (int j = 0; j < 5; ++j) v += xv[4 + i] * sv[4 + j] * cg[OFF221 + (i * 5 + j) * 3 + k]; T[25 + k] = v;
      v = 0.f; for (int i = 0; i < 7; ++i) for (int j = 0; j < 7; ++j) v += xv[9 + i] * sv[9 + j] * cg[OFF331 + (i * 7 + j) * 3 + k]; T[28 + k] = v;
    }
  }
  __syncthreads();

  // combine with per-edge weights, scatter to x1[dst]
  int e = tid >> 3, s = tid & 7;
  int ee = e0 + e;
  bool valid = ee < E;
  int dst = edst[valid ? ee : (E - 1)];
  const float* W = &sWc[e][0];
  const float* T = &sT[e][0];
  if (valid) {
    // scalar outputs: slot0 (16 @ off 0), slot2 (8 @ off 32), slot4 (8 @ off 48)
    for (int w = s; w < 16; w += 8) {
      float v = 0.5f * (T[0] * W[w] + T[1] * W[16 + w] + T[2] * W[32 + w] + T[3] * W[48 + w]);
      atomicAdd(&x1[(size_t)dst * 160 + w], v);
    }
    {
      int w = s;
      float v2 = 0.5f * (T[0] * W[64 + w] + T[1] * W[72 + w] + T[2] * W[80 + w] + T[3] * W[88 + w]);
      atomicAdd(&x1[(size_t)dst * 160 + 32 + w], v2);
      float v4 = 0.5f * (T[0] * W[96 + w] + T[1] * W[104 + w] + T[2] * W[112 + w] + T[3] * W[120 + w]);
      atomicAdd(&x1[(size_t)dst * 160 + 48 + w], v4);
    }
    // 1o outputs: 16 muls x 3 @ off 64 (6 paths, weights 128+16p)
    for (int u = s; u < 16; u += 8)
      for (int k = 0; k < 3; ++k) {
        float v = 0.f;
#pragma unroll
        for (int p = 0; p < 6; ++p) v += T[4 + 3 * p + k] * W[128 + 16 * p + u];
        atomicAdd(&x1[(size_t)dst * 160 + 64 + 3 * u + k], SQRT_HALF * v);
      }
    // 1e outputs: 16 muls x 3 @ off 112 (3 paths, weights 224+16p)
    for (int u = s; u < 16; u += 8)
      for (int k = 0; k < 3; ++k) {
        float v = 0.f;
#pragma unroll
        for (int p = 0; p < 3; ++p) v += T[22 + 3 * p + k] * W[224 + 16 * p + u];
        atomicAdd(&x1[(size_t)dst * 160 + 112 + 3 * u + k], v);
      }
  }
}

// ---------------------------------------------------------------------------
// Gate nonlinearity: x1[n][160] * INV_SQRT_NEIGH -> xg[n][128]
// ---------------------------------------------------------------------------
__global__ __launch_bounds__(256) void gate_kernel(const float* __restrict__ x1,
                                                   float* __restrict__ xg, int N) {
  int idx = blockIdx.x * blockDim.x + threadIdx.x;
  if (idx >= N * 128) return;
  int n = idx >> 7, c = idx & 127;
  const float* r = x1 + (size_t)n * 160;
  float v;
  if (c < 16) {
    v = fmaxf(r[c] * INV_SQRT_NEIGH, 0.0f) * RELU_NORM;
  } else if (c < 32) {
    v = fabsf(r[c] * INV_SQRT_NEIGH);
  } else {
    int u = (c - 32) / 3;
    float gv;
    if (u < 8)        gv = fmaxf(r[32 + u] * INV_SQRT_NEIGH, 0.0f) * RELU_NORM;
    else if (u < 16)  gv = tanhf(r[40 + (u - 8)] * INV_SQRT_NEIGH) * TANH_NORM;
    else if (u < 24)  gv = fmaxf(r[48 + (u - 16)] * INV_SQRT_NEIGH, 0.0f) * RELU_NORM;
    else              gv = tanhf(r[56 + (u - 24)] * INV_SQRT_NEIGH) * TANH_NORM;
    v = r[64 + (c - 32)] * INV_SQRT_NEIGH * gv;
  }
  xg[(size_t)n * 128 + c] = v;
}

// ---------------------------------------------------------------------------
// Fused final layer: radial MLP (WMMA, 304 outputs) -> final TP (xg[src] (x)
// sh -> 7 dims) -> atomic scatter into x7[dst][8(padded)].
// xg (25.6 MB) is read directly from global: it is fully L2-resident (192 MB).
// ---------------------------------------------------------------------------
__global__ __launch_bounds__(256) void final_tp_kernel(
    const float* __restrict__ sh, const float* __restrict__ emb, const float* __restrict__ xg,
    const int* __restrict__ esrc, const int* __restrict__ edst, const float* __restrict__ w1f,
    const _Float16* __restrict__ Bf, const float* __restrict__ cg,
    float* __restrict__ x7, int E) {
  __shared__ float sW1[768];
  __shared__ float sEMB[32][4];
  __shared__ __align__(32) _Float16 hA[32 * 256];
  __shared__ float sSH[32][16];
  __shared__ float sW[32][304];
  __shared__ float sT[32][32];

  int tid = threadIdx.x;
  int e0 = blockIdx.x * 32;

  for (int i = tid; i < 768; i += 256) sW1[i] = w1f[i];
  if (tid < 96) {
    int r = tid / 3, c = tid % 3;
    int ee = e0 + r; if (ee >= E) ee = E - 1;
    sEMB[r][c] = emb[ee * 3 + c];
  }
  for (int i = tid; i < 512; i += 256) {
    int r = i >> 4, c = i & 15;
    int ee = e0 + r; if (ee >= E) ee = E - 1;
    sSH[r][c] = sh[ee * 16 + c];
  }
  __syncthreads();

  for (int i = tid; i < 8192; i += 256) {
    int r = i >> 8, c = i & 255;
    float v = (sEMB[r][0] * sW1[c] + sEMB[r][1] * sW1[256 + c] + sEMB[r][2] * sW1[512 + c]) * INV_SQRT3;
    hA[r * 256 + c] = (_Float16)(fmaxf(v, 0.0f) * RELU_NORM);
  }
  __syncthreads();

  fc_gemm<19, 304>(Bf, hA, &sW[0][0], tid);
  __syncthreads();

  // per-edge (1o x 1o -> 0e) dot vectors via CG(1,1,0), xg gathered from L2
  if (tid < 32) {
    int ee = e0 + tid; if (ee >= E) ee = E - 1;
    const float* g = xg + (size_t)esrc[ee] * 128;
    const float* sv = &sSH[tid][0];
    float* T = &sT[tid][0];
    for (int u = 0; u < 8; ++u) {
      float a2 = 0.f, a3 = 0.f, a4 = 0.f, a5 = 0.f;
      for (int i = 0; i < 3; ++i)
        for (int j = 0; j < 3; ++j) {
          float c110 = cg[OFF110 + i * 3 + j];
          float svj = sv[1 + j] * c110;
          a2 += g[32 + 3 * u + i] * svj;
          a3 += g[56 + 3 * u + i] * svj;
          a4 += g[80 + 3 * u + i] * svj;
          a5 += g[104 + 3 * u + i] * svj;
        }
      T[u] = a2; T[8 + u] = a3; T[16 + u] = a4; T[24 + u] = a5;
    }
  }
  __syncthreads();

  int e = tid >> 3, s = tid & 7;
  int ee = e0 + e;
  bool valid = ee < E;
  int src = esrc[valid ? ee : (E - 1)];
  int dst = edst[valid ? ee : (E - 1)];
  const float* W = &sW[e][0];
  const float* T = &sT[e][0];
  const float cg000 = cg[OFF000];
  if (valid) {
    if (s == 0) {
      // 0o output: paths (3,1,0) + (4,1,0), pw=0.25
      float v = 0.f;
#pragma unroll
      for (int u = 0; u < 8; ++u) v += T[8 + u] * W[u] + T[16 + u] * W[8 + u];
      atomicAdd(&x7[(size_t)dst * 8 + 0], 0.25f * v);
    } else if (s < 7) {
      int w = s - 1;
      const float* g = xg + (size_t)src * 128;
      float sh0 = sSH[e][0] * cg000;
      float v = 0.f;
#pragma unroll
      for (int u = 0; u < 16; ++u)
        v += g[u] * sh0 * W[16 + 6 * u + w] + g[16 + u] * sh0 * W[112 + 6 * u + w];
#pragma unroll
      for (int u = 0; u < 8; ++u)
        v += T[u] * W[208 + 6 * u + w] + T[24 + u] * W[256 + 6 * u + w];
      atomicAdd(&x7[(size_t)dst * 8 + 1 + w], FINAL_B * v);
    }
  }
}

// ---------------------------------------------------------------------------
// Graph readout: out[g][7] += x7[n][c] * INV_SQRT_NEIGH * 0.5
// ---------------------------------------------------------------------------
__global__ __launch_bounds__(256) void reduce_kernel(const float* __restrict__ x7,
                                                     const int* __restrict__ batch,
                                                     float* __restrict__ out, int N) {
  int n = blockIdx.x * blockDim.x + threadIdx.x;
  if (n >= N) return;
  int g = batch[n];
#pragma unroll
  for (int c = 0; c < 7; ++c)
    atomicAdd(&out[g * 7 + c], x7[(size_t)n * 8 + c] * (INV_SQRT_NEIGH * 0.5f));
}

// ---------------------------------------------------------------------------
// Host launcher. Workspace layout (~108.3 MB): sh, emb, x0, x1, xg, x7,
// Bfrag(conv), Bfrag(final), cg tables. All regions 256B aligned.
// ---------------------------------------------------------------------------
extern "C" void kernel_launch(void* const* d_in, const int* in_sizes, int n_in,
                              void* d_out, int out_size, void* d_ws, size_t ws_size,
                              hipStream_t stream) {
  const float* pos  = (const float*)d_in[0];
  const int* batch  = (const int*)d_in[1];
  const int* esrc   = (const int*)d_in[2];
  const int* edst   = (const int*)d_in[3];
  const float* w1c  = (const float*)d_in[4];
  const float* w2c  = (const float*)d_in[5];
  const float* w1f  = (const float*)d_in[6];
  const float* w2f  = (const float*)d_in[7];
  float* out = (float*)d_out;

  const int N = in_sizes[0] / 3;
  const int E = in_sizes[2];

  char* ws = (char*)d_ws;
  auto align256 = [](size_t x) { return (x + 255) & ~(size_t)255; };
  size_t o = 0;
  float* sh_buf  = (float*)(ws + o); o = align256(o + (size_t)E * 16 * 4);
  float* emb_buf = (float*)(ws + o); o = align256(o + (size_t)E * 3 * 4);
  float* x0_buf  = (float*)(ws + o); o = align256(o + (size_t)N * 16 * 4);
  float* x1_buf  = (float*)(ws + o); o = align256(o + (size_t)N * 160 * 4);
  float* xg_buf  = (float*)(ws + o); o = align256(o + (size_t)N * 128 * 4);
  float* x7_buf  = (float*)(ws + o); o = align256(o + (size_t)N * 8 * 4);
  _Float16* Bc   = (_Float16*)(ws + o); o = align256(o + (size_t)17 * 8 * 32 * 16 * 2);
  _Float16* Bf   = (_Float16*)(ws + o); o = align256(o + (size_t)19 * 8 * 32 * 16 * 2);
  float* cg_buf  = (float*)(ws + o); o = align256(o + (size_t)CG_FLOATS * 4);
  (void)ws_size; (void)n_in; (void)out_size;

  // zero atomic accumulation targets
  hipMemsetAsync(x0_buf, 0, (size_t)N * 16 * 4, stream);
  hipMemsetAsync(x1_buf, 0, (size_t)N * 160 * 4, stream);
  hipMemsetAsync(x7_buf, 0, (size_t)N * 8 * 4, stream);
  hipMemsetAsync(out, 0, (size_t)16 * 7 * 4, stream);

  cg_prep_kernel<<<1, 32, 0, stream>>>(cg_buf);
  wprep_kernel<<<(19 * 8 * 32 * 16 + 255) / 256, 256, 0, stream>>>(w2c, w2f, Bc, Bf);
  edge_geom_kernel<<<(E + 255) / 256, 256, 0, stream>>>(pos, esrc, edst, sh_buf, emb_buf, x0_buf, E);
  conv_tp_kernel<<<(E + 31) / 32, 256, 0, stream>>>(sh_buf, emb_buf, x0_buf, esrc, edst,
                                                    w1c, Bc, cg_buf, x1_buf, E);
  gate_kernel<<<((size_t)N * 128 + 255) / 256, 256, 0, stream>>>(x1_buf, xg_buf, N);
  final_tp_kernel<<<(E + 31) / 32, 256, 0, stream>>>(sh_buf, emb_buf, xg_buf, esrc, edst,
                                                     w1f, Bf, cg_buf, x7_buf, E);
  reduce_kernel<<<(N + 255) / 256, 256, 0, stream>>>(x7_buf, batch, out, N);
}
